// PeriodicLoss_diffFs_79259326480951
// MI455X (gfx1250) — matrix operand order repaired and checked by reading, not compiled, not run on hardware
//
#include <hip/hip_runtime.h>

typedef __attribute__((ext_vector_type(16))) _Float16 v16h;
typedef __attribute__((ext_vector_type(2)))  _Float16 h2;
typedef __attribute__((ext_vector_type(8)))  float    v8f;

#define PL_B        32768
#define PL_T        1024
#define PL_PB_LO    (40.0f/60.0f)
#define PL_PB_HI    (180.0f/60.0f)
#define PL_DELTA_HZ (6.0f/60.0f)
#define PL_W0       1.0f
#define PL_W1       1.0f
#define PL_BLOCKS   512
#define PL_TPB      256
#define PL_WAVES    (PL_TPB/32)

static __device__ __forceinline__ v8f wmma_f16(v16h a, v16h b, v8f c) {
  // D = A(16x32 f16) * B(32x16 f16) + C(16x16 f32)
  return __builtin_amdgcn_wmma_f32_16x16x32_f16(false, a, false, b, (short)0, c,
                                                false, false);
}

__global__ void __launch_bounds__(PL_TPB)
__attribute__((amdgpu_waves_per_eu(4)))
periodic_loss_fft_kernel(const float* __restrict__ preds,
                         const float* __restrict__ Fs,
                         float* __restrict__ out)
{
  // Per-wave 32x32 complex (half2) bounce buffer for the inter-stage transpose.
  __shared__ h2 pbuf[PL_WAVES][32 * 32];
  // Block-shared twiddle table: {cos,sin} of -2*pi*k1*n2/1024, idx = k1*32+n2.
  __shared__ float2 twtab[1024];

  const int lane  = threadIdx.x & 31;
  const int wave  = threadIdx.x >> 5;
  const int g     = lane >> 4;    // lane group (0/1)
  const int ln    = lane & 15;
  const int gwave = blockIdx.x * PL_WAVES + wave;
  const int nwav  = gridDim.x * PL_WAVES;   // 4096 -> exactly 8 rows per wave

  const float W32   = -6.28318530717958647692f / 32.0f;
  const float W1024 = -6.28318530717958647692f / 1024.0f;

  // ---- Build the twiddle table once per block ----
  for (int i = threadIdx.x; i < 1024; i += PL_TPB) {
    const int k1 = i >> 5, n2 = i & 31;
    float s, c;
    __sincosf(W1024 * (float)((k1 * n2) & 1023), &s, &c);
    twtab[i] = make_float2(c, s);
  }

  // ---- Stage-1 A operands: F[k1][n1] = e^{-2pi i k1 n1/32} (16x32 tiles) ----
  v16h FrA[2], FiA[2];
  #pragma unroll
  for (int ib = 0; ib < 2; ++ib) {
    const int k1 = 16 * ib + ln;          // A layout: M = lane&15 (both halves)
    #pragma unroll
    for (int e = 0; e < 16; ++e) {
      const int n1 = 16 * g + e;          // A-layout K mapping
      float s, c;
      __sincosf(W32 * (float)((k1 * n1) & 31), &s, &c);
      FrA[ib][e] = (_Float16)c;
      FiA[ib][e] = (_Float16)s;           // sign baked in (s = -sin(2pi k n/32))
    }
  }

  // ---- Stage-2 B operands, k2 in [0,16) only (jb=1 tiles are mirror bins) --
  v16h FrB0, FiB0, FiBn0;
  {
    const int k2 = ln;                    // B layout: N = lane&15
    #pragma unroll
    for (int e = 0; e < 16; ++e) {
      const int n2 = 16 * g + e;          // B-layout K mapping
      float s, c;
      __sincosf(W32 * (float)((n2 * k2) & 31), &s, &c);
      FrB0[e]  = (_Float16)c;
      FiB0[e]  = (_Float16)s;
      FiBn0[e] = (_Float16)(-s);
    }
  }

  h2* P = pbuf[wave];
  __syncthreads();   // twiddle table ready

  for (int row = gwave; row < PL_B; row += nwav) {
    const float* x = preds + (size_t)row * PL_T;

    // Opaque zero (SGPR) regenerated every iteration: the compiler cannot
    // prove the twiddle-table addresses loop-invariant, so the ds_loads stay
    // inside the loop instead of being hoisted into 64 live VGPRs.
    int tw0 = 0;
    asm volatile("" : "+s"(tw0));

    // ---- Load data into stage-1 B operands: M[n1][n2] = x[32*n1+n2] ----
    // Nyquist accumulator: X[512] = sum_n x[n]*(-1)^n; every element this lane
    // loads has the same parity (-1)^(ln&1)  (n = 32*n1 + col, col = 16jb+ln).
    v16h Mb[2];
    float sumx = 0.0f;
    #pragma unroll
    for (int jb = 0; jb < 2; ++jb) {
      const int col = 16 * jb + ln;
      #pragma unroll
      for (int e = 0; e < 16; ++e) {
        const int n1 = 16 * g + e;
        const float xv = __builtin_nontemporal_load(&x[32 * n1 + col]);
        sumx += xv;
        Mb[jb][e] = (_Float16)xv;
      }
    }
    float alt = (ln & 1) ? -sumx : sumx;  // signed partial of X[512]

    // ---- Stage 1 + twiddle + LDS store, one column-half at a time ----
    // (halves Y liveness: only 4 accumulators live at once)
    #pragma unroll
    for (int jb = 0; jb < 2; ++jb) {
      v8f Yr[2], Yi[2];
      #pragma unroll
      for (int ib = 0; ib < 2; ++ib) {
        v8f z = {};
        Yr[ib] = wmma_f16(FrA[ib], Mb[jb], z);
        Yi[ib] = wmma_f16(FiA[ib], Mb[jb], z);
      }
      const int n2 = ln + 16 * jb;              // C layout: col = lane&15
      #pragma unroll
      for (int ib = 0; ib < 2; ++ib) {
        #pragma unroll
        for (int r = 0; r < 8; ++r) {
          const int k1 = r + 8 * g + 16 * ib;   // C layout: row = r + 8g
          const float2 cs = twtab[k1 * 32 + n2 + tw0];   // one ds_load_b64
          const float yr = Yr[ib][r], yi = Yi[ib][r];
          h2 v;
          v.x = (_Float16)(yr * cs.x - yi * cs.y);
          v.y = (_Float16)(yr * cs.y + yi * cs.x);
          P[k1 * 32 + n2] = v;
        }
        // Scheduling fence: cap in-flight twiddle ds_loads at one sub-tile
        // (8 x b64 = 16 VGPRs) instead of the whole row (~64 VGPRs), which
        // was spilling a stage-1 A operand to scratch.
        asm volatile("" ::: "memory");
      }
    }

    __syncthreads();   // all waves run identical trip counts; orders LDS W->R

    // ---- Re-layout P into stage-2 A operands ----
    v16h PrA[2], PiA[2];
    #pragma unroll
    for (int ib = 0; ib < 2; ++ib) {
      const int k1 = 16 * ib + ln;
      #pragma unroll
      for (int e = 0; e < 16; ++e) {
        const int n2 = 16 * g + e;
        h2 v = P[k1 * 32 + n2];
        PrA[ib][e] = v.x;
        PiA[ib][e] = v.y;
      }
    }

    // ---- Stage 2 (k2 < 16 only): Z = P * F (8 WMMA) ----
    v8f Zr[2], Zi[2];
    #pragma unroll
    for (int ib = 0; ib < 2; ++ib) {
      v8f z = {};
      v8f t0 = wmma_f16(PiA[ib], FiBn0, z);    // -Pi*Fi
      Zr[ib] = wmma_f16(PrA[ib], FrB0, t0);    // Pr*Fr - Pi*Fi
      v8f t1 = wmma_f16(PiA[ib], FrB0, z);     //  Pi*Fr
      Zi[ib] = wmma_f16(PrA[ib], FiB0, t1);    // Pr*Fi + Pi*Fr
    }

    // ---- Band edges / delta from Fs (monotone axis -> clamped rounding) ----
    const float fs   = Fs[row];
    const float cfac = fs * 0.5f / 512.0f;           // Hz per bin, (NF-1)=512
    int left, right;
    {
      float kf = PL_PB_LO / cfac;
      int k0 = (int)floorf(kf);
      left = ((kf - (float)k0) <= ((float)(k0 + 1) - kf)) ? k0 : k0 + 1;
      left = min(max(left, 0), 512);
      kf = PL_PB_HI / cfac;
      k0 = (int)floorf(kf);
      right = ((kf - (float)k0) <= ((float)(k0 + 1) - kf)) ? k0 : k0 + 1;
      right = min(max(right, 0), 512);
    }
    const float df = rintf(PL_DELTA_HZ / (fs * 0.5f / 513.0f));  // NF = 513
    const int d = max(1, (int)df);

    // ---- Pass 1: psd, total, in-band sum, in-band argmax (first occurrence)
    // This lane's 16 bins: k = (r + 8g + 16ib) + 32*ln, each k in [0,512) once.
    float psd[2][8];
    float total = 0.0f, inb = 0.0f, maxv = -1.0f;
    int maxk = 1 << 30;
    #pragma unroll
    for (int ib = 0; ib < 2; ++ib)
      #pragma unroll
      for (int r = 0; r < 8; ++r) {
        const int k = (r + 8 * g + 16 * ib) + 32 * ln;
        const float zr = Zr[ib][r], zi = Zi[ib][r];
        const float p = zr * zr + zi * zi;
        psd[ib][r] = p;
        total += p;
        if (k >= left && k < right) {
          inb += p;
          if (p > maxv || (p == maxv && k < maxk)) { maxv = p; maxk = k; }
        }
      }
    #pragma unroll
    for (int off = 16; off > 0; off >>= 1) {
      total += __shfl_xor(total, off, 32);
      inb   += __shfl_xor(inb, off, 32);
      alt   += __shfl_xor(alt, off, 32);
      const float ov = __shfl_xor(maxv, off, 32);
      const int   ok = __shfl_xor(maxk, off, 32);
      if (ov > maxv || (ov == maxv && ok < maxk)) { maxv = ov; maxk = ok; }
    }
    // Nyquist bin k = 512 (real): psd = (sum x[n]*(-1)^n)^2
    const float p512 = alt * alt;
    total += p512;
    if (left <= 512 && 512 < right) {
      inb += p512;
      if (p512 > maxv) { maxv = p512; maxk = 512; }
    }

    // ---- Pass 2: energy in the peak-exclusion window ----
    const int lo = max(left, maxk - d);
    const int hi = min(right, maxk + d);
    float wsum = 0.0f;
    #pragma unroll
    for (int ib = 0; ib < 2; ++ib)
      #pragma unroll
      for (int r = 0; r < 8; ++r) {
        const int k = (r + 8 * g + 16 * ib) + 32 * ln;
        if (k >= lo && k < hi) wsum += psd[ib][r];
      }
    #pragma unroll
    for (int off = 16; off > 0; off >>= 1)
      wsum += __shfl_xor(wsum, off, 32);
    if (512 >= lo && 512 < hi) wsum += p512;

    const float band   = (total - inb) / (1e-8f + total);
    const float sparse = (inb - wsum) / (inb + 1e-8f * (float)(right - left));
    if (lane == 0)
      atomicAdd(out, (PL_W0 * band + PL_W1 * sparse) * (1.0f / (float)PL_B));

    __syncthreads();   // protect LDS before next iteration overwrites it
  }
}

__global__ void pl_zero_kernel(float* __restrict__ out, int n) {
  const int i = blockIdx.x * blockDim.x + threadIdx.x;
  if (i < n) out[i] = 0.0f;
}

extern "C" void kernel_launch(void* const* d_in, const int* in_sizes, int n_in,
                              void* d_out, int out_size, void* d_ws, size_t ws_size,
                              hipStream_t stream) {
  (void)in_sizes; (void)n_in; (void)d_ws; (void)ws_size;
  const float* preds = (const float*)d_in[0];
  const float* fs    = (const float*)d_in[1];
  float* out         = (float*)d_out;

  pl_zero_kernel<<<(out_size + 255) / 256, 256, 0, stream>>>(out, out_size);
  periodic_loss_fft_kernel<<<PL_BLOCKS, PL_TPB, 0, stream>>>(preds, fs, out);
}
